// TransformerEncoder_83665962926232
// MI455X (gfx1250) — matrix operand order repaired and checked by reading, not compiled, not run on hardware
//
#include <hip/hip_runtime.h>

// ---- problem constants (match reference) ----
#define SEQL 2048
#define NB   4
#define NE   256
#define NH   8
#define NHD  32
#define NDFF 1024
#define NLAY 2
#define NTOK (SEQL * NB)   // 8192 tokens
#define LN_EPS 1e-5f

// async global->LDS path (gfx1250): guarded so both toolchains and the host
// pass compile; fallback is plain load + ds_store staging.
#if __has_builtin(__builtin_amdgcn_global_load_async_to_lds_b128) && \
    __has_builtin(__builtin_amdgcn_s_wait_asynccnt)
#define GEMM_ASYNC 1
#else
#define GEMM_ASYNC 0
#endif

typedef __bf16 v16bf __attribute__((ext_vector_type(16)));
typedef __bf16 v8bf  __attribute__((ext_vector_type(8)));
typedef float  v8f   __attribute__((ext_vector_type(8)));
typedef int    v4i   __attribute__((ext_vector_type(4)));

#if GEMM_ASYNC
typedef __attribute__((address_space(1))) v4i* as1_v4i_p;  // global b128 chunk
typedef __attribute__((address_space(3))) v4i* as3_v4i_p;  // LDS b128 chunk
#endif

static __device__ __forceinline__ v16bf cat8(v8bf lo, v8bf hi) {
  return __builtin_shufflevector(lo, hi, 0,1,2,3,4,5,6,7,8,9,10,11,12,13,14,15);
}
static __device__ __forceinline__ v8f wmma_bf16(v16bf a, v16bf b, v8f c) {
  // D(f32 16x16) = A(bf16 16x32) * B(bf16 32x16) + C
  return __builtin_amdgcn_wmma_f32_16x16x32_bf16(false, a, false, b, (short)0, c, false, false);
}
// A-operand: per-lane row m=lane&15, kh=lane>>4: elems K = kh*16 + 0..15 (contiguous).
// `row` must already point at (row m, k0).
static __device__ __forceinline__ v16bf ldA16(const __bf16* row, int kh) {
  return cat8(*(const v8bf*)(row + kh * 16), *(const v8bf*)(row + kh * 16 + 8));
}
// B-operand: per-lane col n=lane&15, kh=lane>>4: elems K = kh*16 + 0..15 (contiguous).
// `p` must already point at (col n, k0 + kh*16).
static __device__ __forceinline__ v16bf ldB16(const __bf16* p) {
  return cat8(*(const v8bf*)p, *(const v8bf*)(p + 8));
}

// -------------------- elementwise f32 -> bf16 --------------------
__global__ void cvt_f32_bf16(const float* __restrict__ in, __bf16* __restrict__ out, int n) {
  int i = blockIdx.x * blockDim.x + threadIdx.x;
  if (i < n) out[i] = (__bf16)in[i];
}

// ---- cooperative stage of a 64(N) x 64(K) bf16 W block into LDS ----
// LDS layout: row-major [n][64]; 512 16B chunks, 4 per thread (128 threads).
static __device__ __forceinline__ void stage_w(const __bf16* __restrict__ W, int K,
                                               int nbase, int k0, __bf16* buf, int tid) {
#pragma unroll
  for (int j = 0; j < 4; ++j) {
    const int c  = tid + j * 128;        // 0..511, coalesced within each group
    const int n  = c >> 3;               // 8 chunks per 64-elem row
    const int ko = (c & 7) * 8;
    const __bf16* g = W + (size_t)(nbase + n) * K + k0 + ko;
    __bf16* l = buf + n * 64 + ko;
#if GEMM_ASYNC
    __builtin_amdgcn_global_load_async_to_lds_b128(
        (as1_v4i_p)(size_t)g,   // AS1 shares representation with generic for global
        (as3_v4i_p)l,           // addrspacecast generic -> LDS
        0, 0);
#else
    *(v8bf*)l = *(const v8bf*)g;
#endif
  }
}

// -------------------- generic WMMA GEMM: C[M,N] = A[M,K] . W[N,K]^T + bias --------
// 4 waves/block; block tile 64(M) x 64(N); W staged in LDS (double buffered, shared
// by the 4 waves -> 4x less B-operand fetch traffic), A streamed from global.
template <bool RELU, bool OUT_BF16>
__global__ void gemm_xwT(const __bf16* __restrict__ A, const __bf16* __restrict__ W,
                         const float* __restrict__ bias,
                         float* __restrict__ Cf, __bf16* __restrict__ Cb,
                         int M, int N, int K) {
  __shared__ __align__(16) __bf16 wtile[2][64 * 64];  // 2 x 8KB
  const int tid  = threadIdx.x;
  const int lane = tid & 31;
  const int wave = tid >> 5;
  const int c16  = lane & 15;
  const int kh   = lane >> 4;
  const int mbase = (blockIdx.x * 4 + wave) * 16;
  const int nbase = blockIdx.y * 64;

  v8f acc[4] = {};
  const __bf16* arow = A + (size_t)(mbase + c16) * K;

  stage_w(W, K, nbase, 0, wtile[0], tid);
  int buf = 0;
  for (int k0 = 0; k0 < K; k0 += 64) {
#if GEMM_ASYNC
    __builtin_amdgcn_s_wait_asynccnt(0);   // own async fills done
#endif
    __syncthreads();                       // publish stage; also fences prior reads
    if (k0 + 64 < K) {
      stage_w(W, K, nbase, k0 + 64, wtile[buf ^ 1], tid);
      __builtin_prefetch(arow + k0 + 64, 0, 1);  // global_prefetch_b8 for A stream
    }
#pragma unroll
    for (int ks = 0; ks < 2; ++ks) {
      v16bf a = ldA16(arow + k0 + ks * 32, kh);
      const __bf16* wl = wtile[buf] + (size_t)c16 * 64 + ks * 32 + kh * 16;
#pragma unroll
      for (int t = 0; t < 4; ++t) {
        v16bf b = ldB16(wl + (size_t)(t * 16) * 64);  // ds_load_b128 x2
        acc[t] = wmma_bf16(a, b, acc[t]);
      }
    }
    buf ^= 1;
  }

  const int mrow = mbase + 8 * kh;  // D layout: lanes 16-31 hold rows M=8..15
#pragma unroll
  for (int t = 0; t < 4; ++t) {
    const int col = nbase + t * 16 + c16;
    const float bv = bias[col];
#pragma unroll
    for (int r = 0; r < 8; ++r) {
      float v = acc[t][r] + bv;
      if (RELU) v = fmaxf(v, 0.0f);
      const size_t idx = (size_t)(mrow + r) * N + col;
      if (OUT_BF16) Cb[idx] = (__bf16)v;
      else          Cf[idx] = v;
    }
  }
}

// -------------------- split packed QKV into per-head bf16 layouts --------------------
// qkv: [NTOK, 3E] f32 (token m = l*NB + b)
// qh, kh_: [bh][L][HD] bf16 ; vt: [bh][HD][L] bf16 (transposed for contiguous PV B-operand)
__global__ void split_heads(const float* __restrict__ qkv,
                            __bf16* __restrict__ qh, __bf16* __restrict__ kh_,
                            __bf16* __restrict__ vt) {
  const int m = blockIdx.x;
  const int l = m >> 2;   // NB == 4
  const int b = m & 3;
#pragma unroll
  for (int j = 0; j < 3; ++j) {
    const int f = j * 256 + threadIdx.x;
    const __bf16 bv = (__bf16)qkv[(size_t)m * (3 * NE) + f];
    const int sec = f >> 8;         // 0:q 1:k 2:v
    const int e   = f & 255;
    const int h   = e >> 5;
    const int d   = e & 31;
    const int bh  = b * NH + h;
    if (sec == 0)      qh [((size_t)bh * SEQL + l) * NHD + d] = bv;
    else if (sec == 1) kh_[((size_t)bh * SEQL + l) * NHD + d] = bv;
    else               vt [((size_t)bh * NHD + d) * SEQL + l] = bv;
  }
}

// -------------------- flash attention: per wave one 16-query tile --------------------
// grid: (SEQL/64, NB*NH), block: 128 (4 waves). ctx written as bf16 [NTOK, E].
__global__ void flash_attn(const __bf16* __restrict__ qh, const __bf16* __restrict__ kmat,
                           const __bf16* __restrict__ vt, __bf16* __restrict__ ctxb) {
  __shared__ __align__(16) __bf16 psh[4][16 * 32];
  const int lane = threadIdx.x & 31;
  const int wave = threadIdx.x >> 5;
  const int c16  = lane & 15;
  const int kh   = lane >> 4;
  const int bh   = blockIdx.y;
  const int b    = bh >> 3;   // NH == 8
  const int h    = bh & 7;
  const int mbase = (blockIdx.x * 4 + wave) * 16;
  const float scale = 0.17677669529663687f;  // 1/sqrt(32)

  // Q tile as WMMA A-operand (16 x 32 == 16 x HD, K-dim done in one shot)
  const __bf16* qrow = qh + ((size_t)bh * SEQL + mbase + c16) * NHD;
  const v16bf aQ = ldA16(qrow, kh);

  v8f o0 = {}, o1 = {};
  float mrun[8], lrun[8];
#pragma unroll
  for (int r = 0; r < 8; ++r) { mrun[r] = -1e30f; lrun[r] = 0.0f; }

  __bf16* pp = psh[wave];  // private per wave; same-wave LDS is in-order (no barrier)

  for (int n0 = 0; n0 < SEQL; n0 += 32) {
    // S tiles: keys n0..n0+15 (tile0) and n0+16..n0+31 (tile1)
    const __bf16* kr0 = kmat + ((size_t)bh * SEQL + n0 + c16) * NHD + kh * 16;
    v8f s0 = wmma_bf16(aQ, ldB16(kr0), v8f{});
    v8f s1 = wmma_bf16(aQ, ldB16(kr0 + 16 * NHD), v8f{});

#pragma unroll
    for (int r = 0; r < 8; ++r) {
      const float a0 = s0[r] * scale;
      const float a1 = s1[r] * scale;
      // row max across the 16 lanes of this half-wave (one matrix row per D-layout)
      float t = fmaxf(a0, a1);
      t = fmaxf(t, __shfl_xor(t, 1));
      t = fmaxf(t, __shfl_xor(t, 2));
      t = fmaxf(t, __shfl_xor(t, 4));
      t = fmaxf(t, __shfl_xor(t, 8));
      const float nmax  = fmaxf(mrun[r], t);
      const float alpha = __expf(mrun[r] - nmax);
      mrun[r] = nmax;
      const float p0 = __expf(a0 - nmax);
      const float p1 = __expf(a1 - nmax);
      float rs = p0 + p1;
      rs += __shfl_xor(rs, 1);
      rs += __shfl_xor(rs, 2);
      rs += __shfl_xor(rs, 4);
      rs += __shfl_xor(rs, 8);
      lrun[r] = lrun[r] * alpha + rs;
      o0[r] *= alpha;
      o1[r] *= alpha;
      // stage P (bf16) into LDS in row-major [row][32] for the A-layout reload
      const int row = r + 8 * kh;
      pp[row * 32 + c16]      = (__bf16)p0;
      pp[row * 32 + 16 + c16] = (__bf16)p1;
    }

    // reload P as WMMA A-operand (16 x 32)
    const v16bf aP = ldA16(pp + c16 * 32, kh);
    // V B-operands from transposed V: contiguous along keys
    const __bf16* vr0 = vt + ((size_t)bh * NHD + c16) * SEQL + n0 + kh * 16;
    o0 = wmma_bf16(aP, ldB16(vr0), o0);
    o1 = wmma_bf16(aP, ldB16(vr0 + 16 * SEQL), o1);
  }

  // finalize and scatter ctx as bf16 [l*NB+b, h*32+d]
#pragma unroll
  for (int r = 0; r < 8; ++r) {
    const float inv = 1.0f / lrun[r];
    const int l = mbase + r + 8 * kh;
    const size_t base = ((size_t)l * NB + b) * NE + h * NHD;
    ctxb[base + c16]      = (__bf16)(o0[r] * inv);
    ctxb[base + 16 + c16] = (__bf16)(o1[r] * inv);
  }
}

// -------------------- residual add + LayerNorm (1 wave per token) --------------------
__global__ void add_ln(const float* __restrict__ x, const float* __restrict__ res,
                       const float* __restrict__ sc, const float* __restrict__ bi,
                       float* __restrict__ xout, __bf16* __restrict__ xbout) {
  const int m = blockIdx.x;
  const int lane = threadIdx.x;
  float y[8];
  float s = 0.0f;
#pragma unroll
  for (int j = 0; j < 8; ++j) {
    const int e = lane + 32 * j;
    y[j] = x[(size_t)m * NE + e] + res[(size_t)m * NE + e];
    s += y[j];
  }
#pragma unroll
  for (int mk = 1; mk < 32; mk <<= 1) s += __shfl_xor(s, mk);
  const float mean = s * (1.0f / NE);
  float v = 0.0f;
#pragma unroll
  for (int j = 0; j < 8; ++j) { const float d = y[j] - mean; v += d * d; }
#pragma unroll
  for (int mk = 1; mk < 32; mk <<= 1) v += __shfl_xor(v, mk);
  const float rstd = rsqrtf(v * (1.0f / NE) + LN_EPS);
#pragma unroll
  for (int j = 0; j < 8; ++j) {
    const int e = lane + 32 * j;
    const float o = (y[j] - mean) * rstd * sc[e] + bi[e];
    xout [(size_t)m * NE + e] = o;
    xbout[(size_t)m * NE + e] = (__bf16)o;
  }
}

// ==================== host-side orchestration ====================
extern "C" void kernel_launch(void* const* d_in, const int* in_sizes, int n_in,
                              void* d_out, int out_size, void* d_ws, size_t ws_size,
                              hipStream_t stream) {
  (void)in_sizes; (void)n_in; (void)out_size; (void)ws_size;
  const float* x     = (const float*)d_in[0];
  const float* in_w  = (const float*)d_in[1];
  const float* in_b  = (const float*)d_in[2];
  const float* out_w = (const float*)d_in[3];
  const float* out_b = (const float*)d_in[4];
  const float* ln1s  = (const float*)d_in[5];
  const float* ln1b  = (const float*)d_in[6];
  const float* w1    = (const float*)d_in[7];
  const float* b1    = (const float*)d_in[8];
  const float* w2    = (const float*)d_in[9];
  const float* b2    = (const float*)d_in[10];
  const float* ln2s  = (const float*)d_in[11];
  const float* ln2b  = (const float*)d_in[12];
  float* xout = (float*)d_out;  // running residual stream lives in d_out

  // workspace carve-up (~55 MB); qkv(f32) and FFN-hidden(bf16) share one region
  char* p = (char*)d_ws;
  auto carve = [&](size_t bytes) -> char* {
    char* r = p;
    p += (bytes + 255) & ~(size_t)255;
    return r;
  };
  __bf16* xb    = (__bf16*)carve((size_t)NTOK * NE * 2);           // 4 MB
  char*   big   = carve((size_t)NTOK * 3 * NE * 4);                // 24 MB (union)
  float*  qkv   = (float*)big;
  __bf16* hb    = (__bf16*)big;                                    // FFN hidden (16 MB)
  float*  tmpf  = (float*)carve((size_t)NTOK * NE * 4);            // attn / ff out, 8 MB
  __bf16* qh    = (__bf16*)carve((size_t)NTOK * NE * 2);           // 4 MB
  __bf16* khd   = (__bf16*)carve((size_t)NTOK * NE * 2);           // 4 MB
  __bf16* vt    = (__bf16*)carve((size_t)NTOK * NE * 2);           // 4 MB
  __bf16* ctxb  = (__bf16*)carve((size_t)NTOK * NE * 2);           // 4 MB
  __bf16* in_wb = (__bf16*)carve((size_t)NLAY * 3 * NE * NE * 2);
  __bf16* ouwb  = (__bf16*)carve((size_t)NLAY * NE * NE * 2);
  __bf16* w1b   = (__bf16*)carve((size_t)NLAY * NDFF * NE * 2);
  __bf16* w2b   = (__bf16*)carve((size_t)NLAY * NE * NDFF * 2);

  auto cvt = [&](const float* src, __bf16* dst, int n) {
    cvt_f32_bf16<<<(n + 255) / 256, 256, 0, stream>>>(src, dst, n);
  };
  cvt(x, xb, NTOK * NE);
  cvt(in_w,  in_wb, NLAY * 3 * NE * NE);
  cvt(out_w, ouwb,  NLAY * NE * NE);
  cvt(w1,    w1b,   NLAY * NDFF * NE);
  cvt(w2,    w2b,   NLAY * NE * NDFF);

  for (int i = 0; i < NLAY; ++i) {
    const float* xsrc = (i == 0) ? x : xout;

    // QKV projection: [8192,256] x [768,256]^T
    gemm_xwT<false, false><<<dim3(NTOK / 64, (3 * NE) / 64), 128, 0, stream>>>(
        xb, in_wb + (size_t)i * 3 * NE * NE, in_b + (size_t)i * 3 * NE,
        qkv, nullptr, NTOK, 3 * NE, NE);

    split_heads<<<NTOK, 256, 0, stream>>>(qkv, qh, khd, vt);

    flash_attn<<<dim3(SEQL / 64, NB * NH), 128, 0, stream>>>(qh, khd, vt, ctxb);

    // out projection: [8192,256] x [256,256]^T
    gemm_xwT<false, false><<<dim3(NTOK / 64, NE / 64), 128, 0, stream>>>(
        ctxb, ouwb + (size_t)i * NE * NE, out_b + (size_t)i * NE,
        tmpf, nullptr, NTOK, NE, NE);

    add_ln<<<NTOK, 32, 0, stream>>>(xsrc, tmpf, ln1s + (size_t)i * NE,
                                    ln1b + (size_t)i * NE, xout, xb);

    // FFN up + ReLU: [8192,256] x [1024,256]^T -> bf16
    gemm_xwT<true, true><<<dim3(NTOK / 64, NDFF / 64), 128, 0, stream>>>(
        xb, w1b + (size_t)i * NDFF * NE, b1 + (size_t)i * NDFF,
        nullptr, hb, NTOK, NDFF, NE);

    // FFN down: [8192,1024] x [256,1024]^T -> f32
    gemm_xwT<false, false><<<dim3(NTOK / 64, NE / 64), 128, 0, stream>>>(
        hb, w2b + (size_t)i * NE * NDFF, b2 + (size_t)i * NE,
        tmpf, nullptr, NTOK, NE, NDFF);

    add_ln<<<NTOK, 32, 0, stream>>>(xout, tmpf, ln2s + (size_t)i * NE,
                                    ln2b + (size_t)i * NE, xout, xb);
  }
}
